// GraphEncoder_43190191128720
// MI455X (gfx1250) — matrix operand order repaired and checked by reading, not compiled, not run on hardware
//
#include <hip/hip_runtime.h>
#include <hip/hip_bf16.h>

// ---------------------------------------------------------------------------
// Types for WMMA (CDNA5 / gfx1250, wave32)
// ---------------------------------------------------------------------------
typedef __attribute__((ext_vector_type(16))) __bf16 v16bf;
typedef __attribute__((ext_vector_type(8)))  __bf16 v8bf;
typedef __attribute__((ext_vector_type(8)))  float  v8f;

#define BATCH   128
#define NNODES  1024
#define NEDGE   16384
#define EMBD    100
#define HIDD    100
#define MROWS   (BATCH * NNODES)   // 131072
#define KPAD    128                 // K padded 100 -> 128 (4 x k32 WMMA steps)

__device__ __forceinline__ unsigned short f2bf(float f) {
  unsigned int u = __builtin_bit_cast(unsigned int, f);
  u += 0x7fffu + ((u >> 16) & 1u);           // round-to-nearest-even
  return (unsigned short)(u >> 16);
}
__device__ __forceinline__ v16bf cat16(v8bf lo, v8bf hi) {
  return __builtin_shufflevector(lo, hi, 0,1,2,3,4,5,6,7,8,9,10,11,12,13,14,15);
}

// Branch-free activations on the native trans path (v_exp_f32 + v_rcp_f32).
// sigmoid(x) = rcp(1 + 2^(-x*log2e));  tanh(x) = (t-1)*rcp(t+1), t = 2^(2x*log2e)
__device__ __forceinline__ float fast_sigmoid(float x) {
  float t = __builtin_amdgcn_exp2f(x * -1.44269504088896340736f);
  return __builtin_amdgcn_rcpf(1.0f + t);
}
__device__ __forceinline__ float fast_tanh(float x) {
  x = fminf(fmaxf(x, -9.0f), 9.0f);          // keep 2^(2x*log2e) finite
  float t = __builtin_amdgcn_exp2f(x * 2.88539008177792681472f);
  return (t - 1.0f) * __builtin_amdgcn_rcpf(t + 1.0f);
}

// ---------------------------------------------------------------------------
// Pack rows of a f32 [*,100] matrix (optionally gathered through an index
// array) into bf16 [*,128] (K zero-padded).  One wave per row.
// ---------------------------------------------------------------------------
__global__ void pack_rows_bf16(const float* __restrict__ src,
                               const int* __restrict__ gather_idx,
                               unsigned short* __restrict__ dst, int nrows) {
  int wid  = (blockIdx.x * blockDim.x + threadIdx.x) >> 5;
  int lane = threadIdx.x & 31;
  if (wid >= nrows) return;
  long srow = gather_idx ? (long)gather_idx[wid] : (long)wid;
  const float* s = src + srow * EMBD;
  unsigned short* d = dst + (long)wid * KPAD;
#pragma unroll
  for (int i = 0; i < 4; i++) {
    int k = i * 32 + lane;
    float v = (k < EMBD) ? s[k] : 0.0f;
    d[k] = f2bf(v);
  }
}

// ---------------------------------------------------------------------------
// Pack the three weight matrices into bf16 B^T layout: BT[n][k], K padded 128.
//   gcn_wt : 112x128  from gcn_w[k*100+n]   (out = X @ W)
//   wih_t  : 304x128  from w_ih[n*100+k]    (out = X @ W_ih^T, already [n][k])
//   whh_t  : 304x128  from w_hh[n*100+k]
// ---------------------------------------------------------------------------
__global__ void pack_weights(const float* __restrict__ gcn_w,
                             const float* __restrict__ w_ih,
                             const float* __restrict__ w_hh,
                             unsigned short* __restrict__ gcn_wt,
                             unsigned short* __restrict__ wih_t,
                             unsigned short* __restrict__ whh_t) {
  int i = blockIdx.x * blockDim.x + threadIdx.x;
  const int S0 = 112 * 128, S1 = 304 * 128;
  if (i < S0) {
    int n = i >> 7, k = i & 127;
    float v = (n < 100 && k < 100) ? gcn_w[k * 100 + n] : 0.0f;
    gcn_wt[i] = f2bf(v);
  } else if (i < S0 + S1) {
    int j = i - S0; int n = j >> 7, k = j & 127;
    float v = (n < 300 && k < 100) ? w_ih[n * 100 + k] : 0.0f;
    wih_t[j] = f2bf(v);
  } else if (i < S0 + 2 * S1) {
    int j = i - S0 - S1; int n = j >> 7, k = j & 127;
    float v = (n < 300 && k < 100) ? w_hh[n * 100 + k] : 0.0f;
    whh_t[j] = f2bf(v);
  }
}

// ---------------------------------------------------------------------------
// bf16 WMMA GEMM with A-reuse:  C[M,NREAL] f32 = A[M,128]bf16 @ BT^T (+bias).
// Each wave owns one 16-row M-tile: loads its 4 A fragments ONCE into
// registers, then loops over all N-tiles (B fragments stream from the
// L2-resident packed weights).  grid = M/128, block = 256 (8 waves).
// ---------------------------------------------------------------------------
template <int NREAL, int NTILES>
__global__ void __launch_bounds__(256)
gemm_bf16(const unsigned short* __restrict__ A,
          const unsigned short* __restrict__ BT,
          const float* __restrict__ bias,     // may be null
          float* __restrict__ C) {
  int lane = threadIdx.x & 31;
  int wid  = threadIdx.x >> 5;
  long mtile = (long)blockIdx.x * 8 + wid;

  const unsigned short* arow = A + (mtile * 16 + (lane & 15)) * KPAD;
  int akoff = (lane < 16) ? 0 : 8;    // ISA A-layout: lanes 16-31 hold K+8
  int bkoff = (lane < 16) ? 0 : 8;    // B-layout matches A: lanes 16-31 hold K+8
  int mbase = (lane < 16) ? 0 : 8;

  v16bf af[4];
#pragma unroll
  for (int kb = 0; kb < 4; kb++) {
    const unsigned short* p = arow + kb * 32 + akoff;
    af[kb] = cat16(*reinterpret_cast<const v8bf*>(p),
                   *reinterpret_cast<const v8bf*>(p + 16));
  }

#pragma unroll
  for (int nt = 0; nt < NTILES; nt++) {
    int n = nt * 16 + (lane & 15);
    float bv = (bias && n < NREAL) ? bias[n] : 0.0f;
    v8f acc = {bv, bv, bv, bv, bv, bv, bv, bv};
    const unsigned short* brow = BT + (long)n * KPAD;
#pragma unroll
    for (int kb = 0; kb < 4; kb++) {
      const unsigned short* q = brow + kb * 32 + bkoff;
      acc = __builtin_amdgcn_wmma_f32_16x16x32_bf16(
          false, af[kb], false,
          cat16(*reinterpret_cast<const v8bf*>(q),
                *reinterpret_cast<const v8bf*>(q + 16)),
          (short)0, acc, false, false);
    }
    if (n < NREAL) {
#pragma unroll
      for (int v = 0; v < 8; v++) {
        long m = mtile * 16 + mbase + v;
        C[m * NREAL + n] = acc[v];
      }
    }
  }
}

// ---------------------------------------------------------------------------
// seq init: seq[b][n][f] = gcn_b[f]
// ---------------------------------------------------------------------------
__global__ void init_seq(float* __restrict__ seq, const float* __restrict__ gcn_b) {
  long i = (long)blockIdx.x * blockDim.x + threadIdx.x;
  long total = (long)MROWS * HIDD;
  if (i < total) seq[i] = gcn_b[i % HIDD];
}

// ---------------------------------------------------------------------------
// SpMM scatter: seq[b,row,:] += val * support[b,col,:]   (one wave per edge)
// f32 global atomics -> global_atomic_add_f32 (no-return).
// ---------------------------------------------------------------------------
__global__ void __launch_bounds__(256)
spmm_scatter(const int* __restrict__ rows, const int* __restrict__ cols,
             const float* __restrict__ vals,
             const float* __restrict__ support, float* __restrict__ seq) {
  long wid  = ((long)blockIdx.x * blockDim.x + threadIdx.x) >> 5;
  int  lane = threadIdx.x & 31;
  if (wid >= (long)BATCH * NEDGE) return;
  long b = wid / NEDGE;
  long e = wid - b * NEDGE;
  int  r = rows[b * NEDGE + e];
  int  c = cols[b * NEDGE + e];
  float v = vals[b * NEDGE + e];
  const float* sp = support + (b * NNODES + c) * HIDD;
  float*       dp = seq     + (b * NNODES + r) * HIDD;
#pragma unroll
  for (int i = 0; i < 4; i++) {
    int f = i * 32 + lane;
    if (f < HIDD) atomicAdd(&dp[f], v * sp[f]);
  }
}

// ---------------------------------------------------------------------------
// GRU recurrence + final FC, fused.  8 workgroups x 16 batch rows.
// Block = 608 threads = 19 waves: wave w owns N-tile w of gh = h @ w_hh^T,
// so the per-step WMMA chain is only 4 deep.  w_hh fragments (+b_hh folded
// into acc init) are pinned in registers across all 1024 steps; the h operand
// (bf16, LDS) is preloaded into registers before the WMMA chain so only one
// dscnt wait is paid.  gx gate inputs prefetch under the WMMAs.  Activations
// are branch-free v_exp_f32/v_rcp_f32 sequences (no IEEE divide, no tanhf
// range-check divergence on the serial critical path).
// ---------------------------------------------------------------------------
__global__ void __launch_bounds__(608)
gru_fc(const float* __restrict__ gx,
       const unsigned short* __restrict__ whh_t,
       const float* __restrict__ b_hh,
       const float* __restrict__ fc1_w, const float* __restrict__ fc1_b,
       float* __restrict__ out) {
  __shared__ __align__(16) unsigned short s_h[16 * 128];  // bf16 matmul operand
  __shared__ float s_gh[16 * 304];                         // gh + b_hh
  __shared__ float s_hf[16 * 100];                         // exact f32 h

  int tid = threadIdx.x, lane = tid & 31, wid = tid >> 5;  // wid = 0..18
  for (int i = tid; i < 16 * 128; i += 608) s_h[i]  = 0;
  for (int i = tid; i < 16 * 100; i += 608) s_hf[i] = 0.0f;

  int akoff = (lane < 16) ? 0 : 8;
  int bkoff = (lane < 16) ? 0 : 8;
  int arow  = lane & 15;
  int mbase = (lane < 16) ? 0 : 8;
  int n     = wid * 16 + (lane & 15);          // this wave's output column

  // Pin this wave's w_hh fragments in registers for the whole recurrence.
  v16bf bfrag[4];
  {
    const unsigned short* brow = whh_t + (long)n * 128;
#pragma unroll
    for (int kb = 0; kb < 4; kb++) {
      const unsigned short* q = brow + kb * 32 + bkoff;
      bfrag[kb] = cat16(*reinterpret_cast<const v8bf*>(q),
                        *reinterpret_cast<const v8bf*>(q + 16));
    }
  }
  float bv = (n < 300) ? b_hh[n] : 0.0f;       // folded into acc init

  long rowbase = (long)blockIdx.x * 16;        // first batch row of this WG
  __syncthreads();

  for (int t = 0; t < NNODES; t++) {
    // ---- prefetch gx gate inputs for this step (hides under the WMMAs)
    float pxr[3], pxz[3], pxn[3];
#pragma unroll
    for (int i = 0; i < 3; i++) {
      int idx = tid + i * 608;
      if (idx < 1600) {
        int m = idx / 100, j = idx - m * 100;
        const float* g = gx + ((rowbase + m) * NNODES + t) * 300;
        pxr[i] = g[j]; pxz[i] = g[100 + j]; pxn[i] = g[200 + j];
      }
    }

    // ---- preload all 4 A fragments of h (one dscnt wait), then 4-deep WMMA
    v16bf afr[4];
#pragma unroll
    for (int kb = 0; kb < 4; kb++) {
      const unsigned short* p = s_h + arow * 128 + kb * 32 + akoff;
      afr[kb] = cat16(*reinterpret_cast<const v8bf*>(p),
                      *reinterpret_cast<const v8bf*>(p + 16));
    }
    v8f acc = {bv, bv, bv, bv, bv, bv, bv, bv};
#pragma unroll
    for (int kb = 0; kb < 4; kb++) {
      acc = __builtin_amdgcn_wmma_f32_16x16x32_bf16(
          false, afr[kb], false, bfrag[kb], (short)0, acc, false, false);
    }
#pragma unroll
    for (int v = 0; v < 8; v++) s_gh[(mbase + v) * 304 + n] = acc[v];
    __syncthreads();

    // ---- gate math (PyTorch order r,z,n); gx carries b_ih, s_gh carries b_hh
#pragma unroll
    for (int i = 0; i < 3; i++) {
      int idx = tid + i * 608;
      if (idx < 1600) {
        int m = idx / 100, j = idx - m * 100;
        float r  = fast_sigmoid(pxr[i] + s_gh[m * 304 + j]);
        float z  = fast_sigmoid(pxz[i] + s_gh[m * 304 + 100 + j]);
        float nn = fast_tanh(pxn[i] + r * s_gh[m * 304 + 200 + j]);
        float hnew = (1.0f - z) * nn + z * s_hf[idx];
        s_hf[idx] = hnew;
        s_h[m * 128 + j] = f2bf(hnew);   // K-pad region [100,128) stays zero
      }
    }
    __syncthreads();
  }

  // ---- fused final FC: out = relu(h @ fc1_w^T + fc1_b)
  for (int idx = tid; idx < 16 * 100; idx += 608) {
    int m = idx / 100, jo = idx - m * 100;
    float a = fc1_b[jo];
    const float* w = fc1_w + jo * 100;
    const float* h = s_hf + m * 100;
#pragma unroll 4
    for (int k = 0; k < 100; k++) a = fmaf(h[k], w[k], a);
    out[(rowbase + m) * 100 + jo] = fmaxf(a, 0.0f);
  }
}

// ---------------------------------------------------------------------------
// Launch graph.  Scratch layout (bytes, total ~191 MB):
//   [0, 32M)          packA bf16 [131072,128]   (reused: emb-gather A, seq A)
//   [32M, 32M+150M)   f32 region: support@+0 (50M) & seq@+50M alive together
//                     during SpMM; gx@+0 (150M) overwrites both afterwards
//   tail              packed bf16 weights (~180 KB)
// ---------------------------------------------------------------------------
extern "C" void kernel_launch(void* const* d_in, const int* in_sizes, int n_in,
                              void* d_out, int out_size, void* d_ws, size_t ws_size,
                              hipStream_t stream) {
  const int*   neighbors = (const int*)d_in[0];
  const int*   adj_row   = (const int*)d_in[1];
  const int*   adj_col   = (const int*)d_in[2];
  const float* adj_val   = (const float*)d_in[3];
  const float* emb_table = (const float*)d_in[4];
  const float* gcn_w     = (const float*)d_in[5];
  const float* gcn_b     = (const float*)d_in[6];
  const float* w_ih      = (const float*)d_in[7];
  const float* w_hh      = (const float*)d_in[8];
  const float* b_ih      = (const float*)d_in[9];
  const float* b_hh      = (const float*)d_in[10];
  const float* fc1_w     = (const float*)d_in[11];
  const float* fc1_b     = (const float*)d_in[12];
  float* out = (float*)d_out;

  char* ws = (char*)d_ws;
  const size_t OFF_PACKA = 0;
  const size_t SZ_PACKA  = (size_t)MROWS * KPAD * 2;           // 33,554,432
  const size_t OFF_BIGF  = OFF_PACKA + SZ_PACKA;
  const size_t SZ_BIGF   = (size_t)MROWS * 300 * 4;            // 157,286,400
  const size_t OFF_WTS   = OFF_BIGF + SZ_BIGF;

  unsigned short* packA = (unsigned short*)(ws + OFF_PACKA);
  float* bigF    = (float*)(ws + OFF_BIGF);
  float* support = bigF;                                        // [131072,100]
  float* seq     = bigF + (size_t)MROWS * HIDD;                 // [131072,100]
  float* gx      = bigF;                                        // [131072,300]
  unsigned short* gcn_wt = (unsigned short*)(ws + OFF_WTS);     // 112*128
  unsigned short* wih_t  = gcn_wt + 112 * 128;                  // 304*128
  unsigned short* whh_t  = wih_t  + 304 * 128;                  // 304*128

  // 1) embedding gather -> bf16 A
  pack_rows_bf16<<<MROWS / 8, 256, 0, stream>>>(emb_table, neighbors, packA, MROWS);
  // 2) weights -> bf16 B^T
  pack_weights<<<(112 * 128 + 2 * 304 * 128 + 255) / 256, 256, 0, stream>>>(
      gcn_w, w_ih, w_hh, gcn_wt, wih_t, whh_t);
  // 3) support = X @ gcn_w
  gemm_bf16<HIDD, 7><<<MROWS / 128, 256, 0, stream>>>(packA, gcn_wt, nullptr, support);
  // 4) seq = gcn_b ; 5) seq += spmm(adj, support)
  init_seq<<<(int)(((long)MROWS * HIDD + 255) / 256), 256, 0, stream>>>(seq, gcn_b);
  spmm_scatter<<<(int)(((long)BATCH * NEDGE * 32) / 256), 256, 0, stream>>>(
      adj_row, adj_col, adj_val, support, seq);
  // 6) seq -> bf16 A (reuse packA)
  pack_rows_bf16<<<MROWS / 8, 256, 0, stream>>>(seq, nullptr, packA, MROWS);
  // 7) gx = seq @ w_ih^T + b_ih  (hoisted non-recurrent GRU gates)
  gemm_bf16<300, 19><<<MROWS / 128, 256, 0, stream>>>(packA, wih_t, b_ih, gx);
  // 8) GRU recurrence (WMMA, w_hh pinned in VGPRs) + fused final FC+ReLU
  gru_fc<<<BATCH / 16, 608, 0, stream>>>(gx, whh_t, b_hh, fc1_w, fc1_b, out);
}
